// CausalSelfAttention_61186104099378
// MI455X (gfx1250) — matrix operand order repaired and checked by reading, not compiled, not run on hardware
//
#include <hip/hip_runtime.h>
#include <hip/hip_bf16.h>

typedef __bf16 bf16_t;
typedef __attribute__((ext_vector_type(16))) __bf16 v16bf;
typedef __attribute__((ext_vector_type(8)))  __bf16 v8bf;
typedef __attribute__((ext_vector_type(8)))  float  v8f;
typedef __attribute__((ext_vector_type(4))) unsigned tdm_u4;
typedef __attribute__((ext_vector_type(8))) int      tdm_i8;
typedef __attribute__((ext_vector_type(4))) int      tdm_i4;

#define D_MODEL   1024
#define NUM_HEADS 16
#define DKH       64
#define BATCH     2
#define SEQ       2048
#define MROWS     (BATCH * SEQ)   // 4096

union FragBF { v16bf v; v8bf h[2]; };

static __device__ inline v8f v8f_zero() {
  v8f z;
#pragma unroll
  for (int i = 0; i < 8; ++i) z[i] = 0.0f;
  return z;
}

// ---------------------------------------------------------------------------
// Tensor Data Mover (gfx1250). Descriptor layout per CDNA5 ISA ch.8.
// ---------------------------------------------------------------------------
#if defined(__HIP_DEVICE_COMPILE__) && __has_builtin(__builtin_amdgcn_tensor_load_to_lds)
#define HAVE_TDM 1
#else
#define HAVE_TDM 0
#endif

#if HAVE_TDM
static __device__ inline unsigned lds_off_b(void* p) {
  return (unsigned)(__SIZE_TYPE__)(__attribute__((address_space(3))) void*)p;
}

// 2D tile load: tile_w bf16 elements per row, tile_h rows, row stride in
// elements. tensor dims == tile dims (tiles are always fully in-bounds here).
static __device__ inline void tdm_load_2d(const void* gsrc, unsigned lds_byte,
                                          unsigned tile_w, unsigned tile_h,
                                          unsigned row_stride_elems) {
  unsigned long long ga = (unsigned long long)(__SIZE_TYPE__)gsrc;
  tdm_u4 g0;
  g0[0] = 1u;                                               // count=1, user mode
  g0[1] = lds_byte;                                         // lds_addr (bytes)
  g0[2] = (unsigned)ga;                                     // global_addr[31:0]
  g0[3] = (unsigned)((ga >> 32) & 0x01ffffffu) | 0x80000000u; // [56:32] | type=2
  tdm_i8 g1;
  g1[0] = (int)(1u << 16);                                  // data_size=1 (2B)
  g1[1] = (int)((tile_w & 0xffffu) << 16);                  // tensor_dim0[15:0]
  g1[2] = (int)(((tile_w >> 16) & 0xffffu) |                // tensor_dim0[31:16]
                ((tile_h & 0xffffu) << 16));                // tensor_dim1[15:0]
  g1[3] = (int)(((tile_h >> 16) & 0xffffu) |                // tensor_dim1[31:16]
                ((tile_w & 0xffffu) << 16));                // tile_dim0
  g1[4] = (int)(tile_h & 0xffffu);                          // tile_dim1 (dim2=0)
  g1[5] = (int)row_stride_elems;                            // dim0_stride[31:0]
  g1[6] = 0;                                                // dim0_stride[47:32]
  g1[7] = 0;
  tdm_i4 z4 = {0, 0, 0, 0};
#if __clang_major__ >= 23
  tdm_i8 z8 = {0, 0, 0, 0, 0, 0, 0, 0};
  __builtin_amdgcn_tensor_load_to_lds(g0, g1, z4, z4, z8, 0);
#else
  __builtin_amdgcn_tensor_load_to_lds(g0, g1, z4, z4, 0);
#endif
}

#if __has_builtin(__builtin_amdgcn_s_wait_tensorcnt)
#define WAIT_TENSOR(n) __builtin_amdgcn_s_wait_tensorcnt(n)
#else
#define WAIT_TENSOR(n) asm volatile("s_wait_tensorcnt " #n ::: "memory")
#endif
#endif  // HAVE_TDM

// ---------------------------------------------------------------------------
// fp32 -> bf16 conversion (grid-stride)
// ---------------------------------------------------------------------------
__global__ void cvt_f32_bf16(const float* __restrict__ src, bf16_t* __restrict__ dst, int n) {
  int i = blockIdx.x * blockDim.x + threadIdx.x;
  int stride = gridDim.x * blockDim.x;
  for (; i < n; i += stride) dst[i] = (bf16_t)src[i];
}

// fp32 (K x N) -> bf16 transposed (N x K): dst[n*K + k] = src[k*N + n]
__global__ void cvt_transpose_bf16(const float* __restrict__ src, bf16_t* __restrict__ dst) {
  int i = blockIdx.x * blockDim.x + threadIdx.x;
  int stride = gridDim.x * blockDim.x;
  const int total = D_MODEL * D_MODEL;
  for (; i < total; i += stride) {
    const int n = i / D_MODEL, k = i % D_MODEL;
    dst[i] = (bf16_t)src[(size_t)k * D_MODEL + n];
  }
}

// ---------------------------------------------------------------------------
// Tiled bf16 GEMM: C(4096x1024) = A(4096x1024) * Bt^T + bias, Bt is [n][k].
// Block 128 threads (4 waves), 128x128 tile, 64x64 per wave, K-stage 64,
// double-buffered TDM pipeline (DMA of stage i+1 overlaps WMMA of stage i).
// MODE 0: fp32 row-major.  MODE 1: bf16 head-major (B,H,T,DKH).
// MODE 2: bf16 transposed head-major (B,H,DKH,SEQ)  [for V].
// ---------------------------------------------------------------------------
template <int MODE>
__global__ __launch_bounds__(128) void gemm4096(const bf16_t* __restrict__ A,
                                                const bf16_t* __restrict__ Bt,
                                                const float* __restrict__ bias,
                                                void* __restrict__ outp) {
  __shared__ __align__(16) bf16_t ldsA[2][128 * 64];  // [m][k], 16KB each
  __shared__ __align__(16) bf16_t ldsB[2][128 * 64];  // [n][k], 16KB each

  const int tid  = threadIdx.x;
  const int wave = tid >> 5;
  const int lane = tid & 31;
  const int hl   = lane >> 4;
  const int l16  = lane & 15;
  const int m0 = blockIdx.y * 128;
  const int n0 = blockIdx.x * 128;
  const int wm = (wave & 1) * 64;
  const int wn = (wave >> 1) * 64;
  const int NK = D_MODEL / 64;  // 16 stages

  v8f acc[4][4];
#pragma unroll
  for (int i = 0; i < 4; ++i)
#pragma unroll
    for (int j = 0; j < 4; ++j) acc[i][j] = v8f_zero();

#if HAVE_TDM
  if (wave == 0) {
    tdm_load_2d(&A[(size_t)m0 * D_MODEL],  lds_off_b(ldsA[0]), 64, 128, D_MODEL);
    tdm_load_2d(&Bt[(size_t)n0 * D_MODEL], lds_off_b(ldsB[0]), 64, 128, D_MODEL);
  }
#endif

  for (int ki = 0; ki < NK; ++ki) {
    const int buf = ki & 1;
#if HAVE_TDM
    if (wave == 0) {
      if (ki + 1 < NK) {
        tdm_load_2d(&A[(size_t)m0 * D_MODEL + (ki + 1) * 64],
                    lds_off_b(ldsA[buf ^ 1]), 64, 128, D_MODEL);
        tdm_load_2d(&Bt[(size_t)n0 * D_MODEL + (ki + 1) * 64],
                    lds_off_b(ldsB[buf ^ 1]), 64, 128, D_MODEL);
        WAIT_TENSOR(2);   // stage ki's two loads (older) have completed
      } else {
        WAIT_TENSOR(0);
      }
    }
#else
    for (int c = tid; c < 1024; c += 128) {
      const int row = c >> 3, col = (c & 7) * 8;
      *(v8bf*)&ldsA[buf][row * 64 + col] =
          *(const v8bf*)&A[(size_t)(m0 + row) * D_MODEL + ki * 64 + col];
      *(v8bf*)&ldsB[buf][row * 64 + col] =
          *(const v8bf*)&Bt[(size_t)(n0 + row) * D_MODEL + ki * 64 + col];
    }
#endif
    __syncthreads();  // stage ki resident in buf

#pragma unroll
    for (int kc = 0; kc < 2; ++kc) {
      FragBF af[4], bfr[4];
#pragma unroll
      for (int mi = 0; mi < 4; ++mi) {
        const bf16_t* p = &ldsA[buf][(wm + mi * 16 + l16) * 64 + kc * 32 + hl * 8];
        af[mi].h[0] = *(const v8bf*)p;         // K {0..7}/{8..15}
        af[mi].h[1] = *(const v8bf*)(p + 16);  // K {16..23}/{24..31}
      }
#pragma unroll
      for (int ni = 0; ni < 4; ++ni) {
        const bf16_t* p = &ldsB[buf][(wn + ni * 16 + l16) * 64 + kc * 32 + hl * 8];
        bfr[ni].h[0] = *(const v8bf*)p;        // K {0..7}/{8..15}
        bfr[ni].h[1] = *(const v8bf*)(p + 16); // K {16..23}/{24..31}
      }
#pragma unroll
      for (int mi = 0; mi < 4; ++mi)
#pragma unroll
        for (int ni = 0; ni < 4; ++ni)
          acc[mi][ni] = __builtin_amdgcn_wmma_f32_16x16x32_bf16(
              false, af[mi].v, false, bfr[ni].v, (short)0, acc[mi][ni], false, false);
    }
    __syncthreads();  // buf free for stage ki+2's DMA
  }

  // Epilogue
#pragma unroll
  for (int mi = 0; mi < 4; ++mi) {
#pragma unroll
    for (int ni = 0; ni < 4; ++ni) {
      const int coln = n0 + wn + ni * 16 + l16;
      const float bv = bias[coln];
#pragma unroll
      for (int r = 0; r < 8; ++r) {
        const int rowm = m0 + wm + mi * 16 + (hl ? 8 + r : r);
        const float val = acc[mi][ni][r] + bv;
        if (MODE == 0) {
          ((float*)outp)[(size_t)rowm * D_MODEL + coln] = val;
        } else {
          const int b = rowm / SEQ, t = rowm % SEQ;
          const int h = coln / DKH, dk = coln % DKH;
          if (MODE == 1) {
            ((bf16_t*)outp)[(((size_t)b * NUM_HEADS + h) * SEQ + t) * DKH + dk] = (bf16_t)val;
          } else {  // MODE 2: V transposed (B,H,DKH,SEQ)
            ((bf16_t*)outp)[(((size_t)b * NUM_HEADS + h) * DKH + dk) * SEQ + t] = (bf16_t)val;
          }
        }
      }
    }
  }
}

// ---------------------------------------------------------------------------
// Flash attention: one (b,h) x 128-query tile per block; 4 waves x 32 queries.
// V pre-transposed in global (B,H,DKH,SEQ). 32-key tiles, online softmax,
// double-buffered TDM staging of K and V^T tiles.
// ---------------------------------------------------------------------------
__global__ __launch_bounds__(128) void flash_attn(const bf16_t* __restrict__ Q,
                                                  const bf16_t* __restrict__ K,
                                                  const bf16_t* __restrict__ Vt,
                                                  bf16_t* __restrict__ O) {
  __shared__ __align__(16) bf16_t ldsK[2][32 * 64];     // [key][d], 4KB each
  __shared__ __align__(16) bf16_t ldsVt[2][64 * 32];    // [d][key], 4KB each
  __shared__ __align__(16) bf16_t ldsP[4][2 * 16 * 32]; // per-wave P tiles, 8KB

  const int tid  = threadIdx.x;
  const int wave = tid >> 5;
  const int lane = tid & 31;
  const int hl   = lane >> 4;
  const int l16  = lane & 15;
  const int bh   = blockIdx.x;  // 0..31
  const int b = bh / NUM_HEADS, h = bh % NUM_HEADS;
  const int qblk  = blockIdx.y;           // 0..15
  const int qbase = qblk * 128 + wave * 32;

  const bf16_t* Qp = Q  + (size_t)bh * SEQ * DKH;
  const bf16_t* Kp = K  + (size_t)bh * SEQ * DKH;
  const bf16_t* Vp = Vt + (size_t)bh * DKH * SEQ;

  // Q A-fragments: 2 q-subtiles x 2 d-steps (constant over key loop)
  FragBF qf[2][2];
#pragma unroll
  for (int qi = 0; qi < 2; ++qi)
#pragma unroll
    for (int s = 0; s < 2; ++s) {
      const bf16_t* p = Qp + (size_t)(qbase + qi * 16 + l16) * DKH + s * 32 + hl * 8;
      qf[qi][s].h[0] = *(const v8bf*)p;
      qf[qi][s].h[1] = *(const v8bf*)(p + 16);
    }

  v8f o[2][4];
#pragma unroll
  for (int qi = 0; qi < 2; ++qi)
#pragma unroll
    for (int c = 0; c < 4; ++c) o[qi][c] = v8f_zero();
  float m[2][8], l[2][8];
  const float NEG = -1e30f;
#pragma unroll
  for (int qi = 0; qi < 2; ++qi)
#pragma unroll
    for (int r = 0; r < 8; ++r) { m[qi][r] = NEG; l[qi][r] = 0.0f; }

  const int ntiles = (qblk + 1) * 4;

#if HAVE_TDM
  if (wave == 0) {
    tdm_load_2d(Kp, lds_off_b(ldsK[0]),  64, 32, DKH);
    tdm_load_2d(Vp, lds_off_b(ldsVt[0]), 32, 64, SEQ);
  }
#endif

  for (int kt = 0; kt < ntiles; ++kt) {
    const int key0 = kt * 32;
    const int buf = kt & 1;
#if HAVE_TDM
    if (wave == 0) {
      if (kt + 1 < ntiles) {
        tdm_load_2d(Kp + (size_t)(key0 + 32) * DKH, lds_off_b(ldsK[buf ^ 1]),  64, 32, DKH);
        tdm_load_2d(Vp + key0 + 32,                 lds_off_b(ldsVt[buf ^ 1]), 32, 64, SEQ);
        WAIT_TENSOR(2);
      } else {
        WAIT_TENSOR(0);
      }
    }
#else
    for (int c = tid; c < 256; c += 128) {
      const int row = c >> 3, col = (c & 7) * 8;
      *(v8bf*)&ldsK[buf][row * 64 + col] =
          *(const v8bf*)&Kp[(size_t)(key0 + row) * DKH + col];
    }
    for (int c = tid; c < 256; c += 128) {
      const int row = c >> 2, col = (c & 3) * 8;
      *(v8bf*)&ldsVt[buf][row * 32 + col] =
          *(const v8bf*)&Vp[(size_t)row * SEQ + key0 + col];
    }
#endif
    __syncthreads();  // tile kt resident in buf

    if (key0 <= qbase + 31) {  // not fully masked for this wave
      // K^T B-fragments: 2 key-halves x 2 d-steps (shared by both q-subtiles)
      FragBF kf[2][2];
#pragma unroll
      for (int kc = 0; kc < 2; ++kc)
#pragma unroll
        for (int s = 0; s < 2; ++s) {
          const bf16_t* p = &ldsK[buf][(kc * 16 + l16) * 64 + s * 32 + hl * 8];
          kf[kc][s].h[0] = *(const v8bf*)p;
          kf[kc][s].h[1] = *(const v8bf*)(p + 16);
        }

      const bool act0 = (key0 <= qbase + 15);
      const float scale = 0.125f;  // 1/sqrt(64)
#pragma unroll
      for (int qi = 0; qi < 2; ++qi) {
        if (qi == 0 && !act0) continue;
        v8f s0 = v8f_zero(), s1 = v8f_zero();
#pragma unroll
        for (int s = 0; s < 2; ++s) {
          s0 = __builtin_amdgcn_wmma_f32_16x16x32_bf16(false, qf[qi][s].v, false, kf[0][s].v,
                                                       (short)0, s0, false, false);
          s1 = __builtin_amdgcn_wmma_f32_16x16x32_bf16(false, qf[qi][s].v, false, kf[1][s].v,
                                                       (short)0, s1, false, false);
        }
        float alpha[8];
#pragma unroll
        for (int r = 0; r < 8; ++r) {
          const int q = qbase + qi * 16 + (hl ? 8 + r : r);
          float v0 = s0[r] * scale; if (key0 + l16 > q)      v0 = NEG;
          float v1 = s1[r] * scale; if (key0 + 16 + l16 > q) v1 = NEG;
          float mx = fmaxf(v0, v1);
          mx = fmaxf(mx, __shfl_xor(mx, 1, 32));
          mx = fmaxf(mx, __shfl_xor(mx, 2, 32));
          mx = fmaxf(mx, __shfl_xor(mx, 4, 32));
          mx = fmaxf(mx, __shfl_xor(mx, 8, 32));
          const float mn = fmaxf(m[qi][r], mx);
          alpha[r] = __expf(m[qi][r] - mn);
          const float p0v = __expf(v0 - mn);
          const float p1v = __expf(v1 - mn);
          float rs = p0v + p1v;
          rs += __shfl_xor(rs, 1, 32);
          rs += __shfl_xor(rs, 2, 32);
          rs += __shfl_xor(rs, 4, 32);
          rs += __shfl_xor(rs, 8, 32);
          l[qi][r] = l[qi][r] * alpha[r] + rs;
          m[qi][r] = mn;
          s0[r] = p0v; s1[r] = p1v;
        }
#pragma unroll
        for (int c = 0; c < 4; ++c)
#pragma unroll
          for (int r = 0; r < 8; ++r) o[qi][c][r] *= alpha[r];

        bf16_t* Pw = &ldsP[wave][qi * 512];
#pragma unroll
        for (int r = 0; r < 8; ++r) {
          const int M = hl ? 8 + r : r;
          Pw[M * 32 + l16]      = (bf16_t)s0[r];
          Pw[M * 32 + 16 + l16] = (bf16_t)s1[r];
        }
      }
      asm volatile("s_wait_dscnt 0" ::: "memory");  // intra-wave LDS RAW fence

      FragBF pf[2];
#pragma unroll
      for (int qi = 0; qi < 2; ++qi) {
        const bf16_t* p = &ldsP[wave][qi * 512 + l16 * 32 + hl * 8];
        pf[qi].h[0] = *(const v8bf*)p;
        pf[qi].h[1] = *(const v8bf*)(p + 16);
      }
#pragma unroll
      for (int c = 0; c < 4; ++c) {
        FragBF vf;
        const bf16_t* p = &ldsVt[buf][(c * 16 + l16) * 32 + hl * 8];
        vf.h[0] = *(const v8bf*)p;
        vf.h[1] = *(const v8bf*)(p + 16);
#pragma unroll
        for (int qi = 0; qi < 2; ++qi) {
          if (qi == 0 && !act0) continue;
          o[qi][c] = __builtin_amdgcn_wmma_f32_16x16x32_bf16(false, pf[qi].v, false, vf.v,
                                                             (short)0, o[qi][c], false, false);
        }
      }
    }
    __syncthreads();  // buf free for tile kt+2's DMA
  }

  // Normalize and write O in (B,T,D) bf16 layout
#pragma unroll
  for (int qi = 0; qi < 2; ++qi)
#pragma unroll
    for (int c = 0; c < 4; ++c)
#pragma unroll
      for (int r = 0; r < 8; ++r) {
        const int q = qbase + qi * 16 + (hl ? 8 + r : r);
        const int d = c * 16 + l16;
        const float val = o[qi][c][r] / l[qi][r];
        O[((size_t)b * SEQ + q) * D_MODEL + h * DKH + d] = (bf16_t)val;
      }
}

// ---------------------------------------------------------------------------
extern "C" void kernel_launch(void* const* d_in, const int* in_sizes, int n_in,
                              void* d_out, int out_size, void* d_ws, size_t ws_size,
                              hipStream_t stream) {
  const float* x  = (const float*)d_in[0];
  // d_in[1] = mask (causality implemented analytically)
  const float* Wq = (const float*)d_in[2];
  const float* bq = (const float*)d_in[3];
  const float* Wk = (const float*)d_in[4];
  const float* bk = (const float*)d_in[5];
  const float* Wv = (const float*)d_in[6];
  const float* bv = (const float*)d_in[7];
  const float* Wo = (const float*)d_in[8];
  const float* bo = (const float*)d_in[9];
  float* out = (float*)d_out;

  // Workspace layout (bytes): 48 MB total
  char* ws = (char*)d_ws;
  bf16_t* xb   = (bf16_t*)(ws);                   // 8 MB  x bf16
  bf16_t* Wqt  = (bf16_t*)(ws + (8ull  << 20));   // 2 MB  Wq^T bf16 [n][k]
  bf16_t* Wkt  = (bf16_t*)(ws + (10ull << 20));   // 2 MB
  bf16_t* Wvt  = (bf16_t*)(ws + (12ull << 20));   // 2 MB
  bf16_t* Wot  = (bf16_t*)(ws + (14ull << 20));   // 2 MB
  bf16_t* Qh   = (bf16_t*)(ws + (16ull << 20));   // 8 MB  (B,H,T,DKH)
  bf16_t* Kh   = (bf16_t*)(ws + (24ull << 20));   // 8 MB  (B,H,T,DKH)
  bf16_t* Vht  = (bf16_t*)(ws + (32ull << 20));   // 8 MB  (B,H,DKH,SEQ)
  bf16_t* Ob   = (bf16_t*)(ws + (40ull << 20));   // 8 MB  (B,T,D)

  cvt_f32_bf16<<<512, 256, 0, stream>>>(x, xb, MROWS * D_MODEL);
  cvt_transpose_bf16<<<256, 256, 0, stream>>>(Wq, Wqt);
  cvt_transpose_bf16<<<256, 256, 0, stream>>>(Wk, Wkt);
  cvt_transpose_bf16<<<256, 256, 0, stream>>>(Wv, Wvt);
  cvt_transpose_bf16<<<256, 256, 0, stream>>>(Wo, Wot);

  dim3 gg(D_MODEL / 128, MROWS / 128);  // (8, 32)
  gemm4096<1><<<gg, 128, 0, stream>>>(xb, Wqt, bq, (void*)Qh);
  gemm4096<1><<<gg, 128, 0, stream>>>(xb, Wkt, bk, (void*)Kh);
  gemm4096<2><<<gg, 128, 0, stream>>>(xb, Wvt, bv, (void*)Vht);

  flash_attn<<<dim3(BATCH * NUM_HEADS, SEQ / 128), 128, 0, stream>>>(Qh, Kh, Vht, Ob);

  gemm4096<0><<<gg, 128, 0, stream>>>(Ob, Wot, bo, (void*)out);
}
